// PINN_38130719654436
// MI455X (gfx1250) — matrix operand order repaired and checked by reading, not compiled, not run on hardware
//
#include <hip/hip_runtime.h>

typedef _Float16 v16h __attribute__((ext_vector_type(16)));
typedef _Float16 v8h  __attribute__((ext_vector_type(8)));
typedef float    v8f  __attribute__((ext_vector_type(8)));

#define NPTS 65536
#define TILE 16          // points per workgroup
#define HID  256
#define ROWB 264         // 256 + 8 halves pad -> 528B (132-dword) row stride: b128 A loads hit
                         // banks 4m..4m+3 per lane m -> conflict-free

// ---------------- init: zero the 8 accumulator floats in workspace ----------------
__global__ void pinn_init_acc(float* acc) {
    if (threadIdx.x < 8) acc[threadIdx.x] = 0.f;
}

// ---------------- prep: swizzle 4 hidden-layer W (f32 [K][N]) into f16 B-fragment layout ----
// Block layout: [layer][n_tile(16)][k_chunk(8)] -> 1024B blocks; lane owns 32B (8 dwords),
// dword v = packed f16 pair {W[k][n], W[k+1][n]}, k = c*32 + (lane>=16)*16 + 2v, n = nt*16 + lane%16.
__global__ void pinn_prep_B(const float* W1, const float* W2, const float* W3,
                            const float* W4, unsigned* wsB) {
    int id = blockIdx.x * blockDim.x + threadIdx.x;      // 131072 total
    const float* Ws[4] = {W1, W2, W3, W4};
    int layer = id >> 15;
    int lane  = (id >> 3) & 31;
    int v     = id & 7;
    int c     = (id >> 8) & 7;
    int nt    = (id >> 11) & 15;
    int hi    = lane >> 4;
    int n     = nt * 16 + (lane & 15);
    int k     = c * 32 + hi * 16 + 2 * v;
    const float* W = Ws[layer];
    union { _Float16 h[2]; unsigned u; } p;
    p.h[0] = (_Float16)W[(size_t)k * HID + n];
    p.h[1] = (_Float16)W[(size_t)(k + 1) * HID + n];
    wsB[id] = p.u;   // output dword index == id by construction
}

// ---------------- main: 16 points/block, 8 waves, 2 N-tiles per wave ------
// Per k-chunk a wave loads ONE A-set (5 derivative streams) and TWO B fragments,
// issuing 10 WMMAs -> halves LDS A-traffic per WMMA vs 1-N-tile mapping.
#define DO_STATE(S, CR0, CR1)                                                          \
    {                                                                                  \
        const _Float16* rp = &sH[S][nl][0];                                            \
        v8h alo = *(const v8h*)(rp + c * 32 + hi * 16);                                \
        v8h ahi = *(const v8h*)(rp + c * 32 + hi * 16 + 8);                            \
        v16h A = __builtin_shufflevector(alo, ahi, 0, 1, 2, 3, 4, 5, 6, 7,             \
                                         8, 9, 10, 11, 12, 13, 14, 15);                \
        CR0 = __builtin_amdgcn_wmma_f32_16x16x32_f16(false, A, false, B0,              \
                                                     (short)0, CR0, false, false);     \
        CR1 = __builtin_amdgcn_wmma_f32_16x16x32_f16(false, A, false, B1,              \
                                                     (short)0, CR1, false, false);     \
    }

__global__ __launch_bounds__(256) void pinn_main(
        const float* __restrict__ vect, const int* __restrict__ vals,
        const float* __restrict__ W0,   const float* __restrict__ b0,
        const float* __restrict__ b1,   const float* __restrict__ b2,
        const float* __restrict__ b3,   const float* __restrict__ b4,
        const float* __restrict__ W5,   const float* __restrict__ b5,
        const unsigned char* __restrict__ wsB, float* __restrict__ acc) {

    __shared__ __align__(16) _Float16 sH[5][TILE][ROWB];  // h, hx, hy, hxx, hyy (f16)
    __shared__ float sred[5][TILE];

    const int tid  = threadIdx.x;
    const int w    = tid >> 5;          // wave id 0..7 -> owns N-tiles 2w, 2w+1
    const int lane = tid & 31;
    const int hi   = lane >> 4;
    const int nl   = lane & 15;
    const int gp0  = blockIdx.x * TILE;

    // ---- layer 0: K=2, scalar. z = x*W0[0][n] + y*W0[1][n] + b0[n]; seed tangents.
#pragma unroll
    for (int i = 0; i < 16; ++i) {
        int e = i * 256 + tid;
        int p = e >> 8, n = e & 255;
        float x  = vect[(size_t)(gp0 + p) * 2 + 0];
        float y  = vect[(size_t)(gp0 + p) * 2 + 1];
        float zx = W0[n], zy = W0[HID + n];
        float z  = x * zx + y * zy + b0[n];
        float t  = tanhf(z), dt = 1.f - t * t;
        sH[0][p][n] = (_Float16)t;
        sH[1][p][n] = (_Float16)(dt * zx);
        sH[2][p][n] = (_Float16)(dt * zy);
        sH[3][p][n] = (_Float16)(-2.f * t * dt * zx * zx);
        sH[4][p][n] = (_Float16)(-2.f * t * dt * zy * zy);
    }
    __syncthreads();

    // ---- layers 1..4: 5 streams x [16,256]x[256,256] via WMMA.
    const float* bs[4] = {b1, b2, b3, b4};
    for (int layer = 0; layer < 4; ++layer) {
        v8f Ca0 = {}, Ca1 = {}, Ca2 = {}, Ca3 = {}, Ca4 = {};   // N-tile 2w
        v8f Cb0 = {}, Cb1 = {}, Cb2 = {}, Cb3 = {}, Cb4 = {};   // N-tile 2w+1
#pragma unroll
        for (int c = 0; c < 8; ++c) {
            const v16h B0 = *(const v16h*)(wsB +
                (((size_t)(layer * 16 + 2 * w + 0) * 8 + c) << 10) + ((size_t)lane << 5));
            const v16h B1 = *(const v16h*)(wsB +
                (((size_t)(layer * 16 + 2 * w + 1) * 8 + c) << 10) + ((size_t)lane << 5));
            DO_STATE(0, Ca0, Cb0)
            DO_STATE(1, Ca1, Cb1)
            DO_STATE(2, Ca2, Cb2)
            DO_STATE(3, Ca3, Cb3)
            DO_STATE(4, Ca4, Cb4)
        }
        float bias0 = bs[layer][32 * w + nl];
        float bias1 = bs[layer][32 * w + 16 + nl];
        __syncthreads();   // all LDS reads done before overwriting states
#pragma unroll
        for (int r = 0; r < 8; ++r) {
            int m  = r + 8 * hi;         // C/D layout: VGPR r, lane-half -> point row
            {   // N-tile 2w
                int n = 32 * w + nl;
                float z  = Ca0[r] + bias0;
                float zx = Ca1[r], zy = Ca2[r], zxx = Ca3[r], zyy = Ca4[r];
                float t  = tanhf(z), dt = 1.f - t * t, c2 = -2.f * t * dt;
                sH[0][m][n] = (_Float16)t;
                sH[1][m][n] = (_Float16)(dt * zx);
                sH[2][m][n] = (_Float16)(dt * zy);
                sH[3][m][n] = (_Float16)(dt * zxx + c2 * zx * zx);
                sH[4][m][n] = (_Float16)(dt * zyy + c2 * zy * zy);
            }
            {   // N-tile 2w+1
                int n = 32 * w + 16 + nl;
                float z  = Cb0[r] + bias1;
                float zx = Cb1[r], zy = Cb2[r], zxx = Cb3[r], zyy = Cb4[r];
                float t  = tanhf(z), dt = 1.f - t * t, c2 = -2.f * t * dt;
                sH[0][m][n] = (_Float16)t;
                sH[1][m][n] = (_Float16)(dt * zx);
                sH[2][m][n] = (_Float16)(dt * zy);
                sH[3][m][n] = (_Float16)(dt * zxx + c2 * zx * zx);
                sH[4][m][n] = (_Float16)(dt * zyy + c2 * zy * zy);
            }
        }
        __syncthreads();
    }

    // ---- layer 5 (linear, N=1): wave w reduces points 2w and 2w+1.
    float ts0 = 0.f, ls0 = 0.f, ts1 = 0.f, ls1 = 0.f;
#pragma unroll
    for (int k8 = 0; k8 < 8; ++k8) {
        int k    = lane + k8 * 32;
        float wh = W5[k];
        int p0 = 2 * w, p1 = 2 * w + 1;
        ts0 += (float)sH[0][p0][k] * wh;
        ls0 += ((float)sH[3][p0][k] + (float)sH[4][p0][k]) * wh;
        ts1 += (float)sH[0][p1][k] * wh;
        ls1 += ((float)sH[3][p1][k] + (float)sH[4][p1][k]) * wh;
    }
#pragma unroll
    for (int off = 16; off > 0; off >>= 1) {
        ts0 += __shfl_down(ts0, off, 32);
        ls0 += __shfl_down(ls0, off, 32);
        ts1 += __shfl_down(ts1, off, 32);
        ls1 += __shfl_down(ls1, off, 32);
    }
    if (lane == 0) {
#pragma unroll
        for (int j = 0; j < 2; ++j) {
            int p   = 2 * w + j;
            float T = (j ? ts1 : ts0) + b5[0];
            float L = (j ? ls1 : ls0);
            int vv  = vals[gp0 + p];
            sred[0][p] = L * L;
            sred[1][p] = (vv == 1) ? (T - 1200.f) * (T - 1200.f) : 0.f;
            sred[2][p] = (vv == 1) ? 1.f : 0.f;
            sred[3][p] = (vv == 2) ? (T - 25.f) * (T - 25.f) : 0.f;
            sred[4][p] = (vv == 2) ? 1.f : 0.f;
        }
    }
    __syncthreads();
    if (tid < 5) {
        float s = 0.f;
#pragma unroll
        for (int i = 0; i < TILE; ++i) s += sred[tid][i];
        atomicAdd(&acc[tid], s);
    }
}

// ---------------- finalize: combine the three loss terms ----------------
__global__ void pinn_finalize(const float* acc, float* out) {
    float cin  = acc[2] < 1.f ? 1.f : acc[2];
    float cout = acc[4] < 1.f ? 1.f : acc[4];
    out[0] = acc[0] / (float)NPTS + acc[1] / cin + acc[3] / cout;
}

extern "C" void kernel_launch(void* const* d_in, const int* in_sizes, int n_in,
                              void* d_out, int out_size, void* d_ws, size_t ws_size,
                              hipStream_t stream) {
    (void)in_sizes; (void)n_in; (void)out_size; (void)ws_size;
    const float* vect = (const float*)d_in[0];
    const int*   vals = (const int*)d_in[1];
    const float* W0 = (const float*)d_in[2];
    const float* b0 = (const float*)d_in[3];
    const float* W1 = (const float*)d_in[4];
    const float* b1 = (const float*)d_in[5];
    const float* W2 = (const float*)d_in[6];
    const float* b2 = (const float*)d_in[7];
    const float* W3 = (const float*)d_in[8];
    const float* b3 = (const float*)d_in[9];
    const float* W4 = (const float*)d_in[10];
    const float* b4 = (const float*)d_in[11];
    const float* W5 = (const float*)d_in[12];
    const float* b5 = (const float*)d_in[13];

    float* acc = (float*)d_ws;                                   // 8 floats
    unsigned char* wsB = (unsigned char*)d_ws + 1024;            // 512 KB swizzled f16 weights

    pinn_init_acc<<<1, 32, 0, stream>>>(acc);
    pinn_prep_B<<<512, 256, 0, stream>>>(W1, W2, W3, W4, (unsigned*)wsB);
    pinn_main<<<NPTS / TILE, 256, 0, stream>>>(vect, vals, W0, b0, b1, b2, b3, b4,
                                               W5, b5, wsB, acc);
    pinn_finalize<<<1, 1, 0, stream>>>(acc, (float*)d_out);
}
